// FHNAttention_86655260165010
// MI455X (gfx1250) — matrix operand order repaired and checked by reading, not compiled, not run on hardware
//
#include <hip/hip_runtime.h>
#include <hip/hip_bf16.h>

typedef __bf16 bf16;
typedef __attribute__((ext_vector_type(16))) __bf16 bf16x16;
typedef __attribute__((ext_vector_type(8)))  __bf16 bf16x8;
typedef __attribute__((ext_vector_type(8)))  float  floatx8;

#define B_    2
#define T_    4096
#define E_    1024
#define H_    16
#define DH_   64
#define CHUNK_ 512

// ---------- WMMA helpers (CDNA5 wave32, 16x16x32 bf16 -> f32) ----------

__device__ __forceinline__ floatx8 wmma_bf16(bf16x16 a, bf16x16 b, floatx8 c) {
  return __builtin_amdgcn_wmma_f32_16x16x32_bf16(false, a, false, b, (short)0, c,
                                                 false, false);
}

__device__ __forceinline__ bf16x16 cat8(bf16x8 lo, bf16x8 hi) {
  return __builtin_shufflevector(lo, hi, 0,1,2,3,4,5,6,7,8,9,10,11,12,13,14,15);
}

// A fragment 16x32 (row-major source, leading dim ld):
// lane L: row = L%16 ; K = 8*(L/16)+{0..7} and 16+8*(L/16)+{0..7}
__device__ __forceinline__ bf16x16 load_afrag(const bf16* base, int ld, int lane) {
  const bf16* p = base + (size_t)(lane & 15) * ld + ((lane >> 4) << 3);
  bf16x8 lo = *(const bf16x8*)(p);
  bf16x8 hi = *(const bf16x8*)(p + 16);
  return cat8(lo, hi);
}

// B fragment 32x16 from memory laid out [n][k] (k contiguous, leading dim ld):
// lane L: col n = L%16 ; K = 8*(L/16)+{0..7} and 16+8*(L/16)+{0..7}
__device__ __forceinline__ bf16x16 load_bfrag(const bf16* base, int ld, int lane) {
  const bf16* p = base + (size_t)(lane & 15) * ld + ((lane >> 4) << 3);
  bf16x8 lo = *(const bf16x8*)(p);
  bf16x8 hi = *(const bf16x8*)(p + 16);
  return cat8(lo, hi);
}

// Row-wise reductions across a 16-lane half (wave32: xor masks 1..8 stay in half)
__device__ __forceinline__ float rmax16(float x) {
#pragma unroll
  for (int off = 1; off < 16; off <<= 1) x = fmaxf(x, __shfl_xor(x, off, 32));
  return x;
}
__device__ __forceinline__ float rsum16(float x) {
#pragma unroll
  for (int off = 1; off < 16; off <<= 1) x += __shfl_xor(x, off, 32);
  return x;
}

// ---------- gfx1250 async global->LDS copy (tracked by ASYNCcnt) ----------
// GV mode: 64-bit global addr in a VGPR pair, per-lane LDS byte address in VDST VGPR.
// Low 32 bits of a generic pointer to LDS are the LDS byte offset (aperture rule).

__device__ __forceinline__ void async_ld_b128(void* lds_dst, const void* gsrc) {
  unsigned lo = (unsigned)(uintptr_t)lds_dst;
  unsigned long long ga = (unsigned long long)(uintptr_t)gsrc;
  asm volatile("global_load_async_to_lds_b128 %0, %1, off"
               :: "v"(lo), "v"(ga) : "memory");
}
__device__ __forceinline__ void wait_async_le4() {
  asm volatile("s_wait_asynccnt 0x4" ::: "memory");
}
__device__ __forceinline__ void wait_async_0() {
  asm volatile("s_wait_asynccnt 0x0" ::: "memory");
}

// ---------- f32 -> bf16 conversion ----------

__global__ __launch_bounds__(256) void k_cvt(const float* __restrict__ src,
                                             bf16* __restrict__ dst, int n) {
  int i = blockIdx.x * 256 + threadIdx.x;
  if (i < n) dst[i] = (bf16)src[i];
}

// ---------- LDS-staged bf16 GEMM core ----------
// C(M x N) = A(M x K, row-major) * B^T where Bmat[n][k] row-major (k contiguous).
// WG tile 128x128, double-buffered 128x32 A and B k-slices in LDS via async copies.
// 8 waves arranged 4(M) x 2(N); each wave computes 32x64 = 2x4 WMMA tiles.

template <typename EPI>
__device__ __forceinline__ void gemm_core(const bf16* __restrict__ A,
                                          const bf16* __restrict__ Bm,
                                          int K, EPI epi) {
  __shared__ __align__(16) bf16 lA[2][128 * 32];
  __shared__ __align__(16) bf16 lB[2][128 * 32];

  const int tid  = threadIdx.x;
  const int lane = tid & 31;
  const int wave = tid >> 5;
  const int tileM = blockIdx.x * 128;
  const int tileN = blockIdx.y * 128;
  const int am = (wave & 3) * 32;   // wave's row offset within tile
  const int bn = (wave >> 2) * 64;  // wave's col offset within tile

  floatx8 acc[2][4];
#pragma unroll
  for (int mt = 0; mt < 2; ++mt)
#pragma unroll
    for (int nt = 0; nt < 4; ++nt)
#pragma unroll
      for (int r = 0; r < 8; ++r) acc[mt][nt][r] = 0.0f;

  const int NK = K / 32;

  // stage loader: 256 threads move 8KB A slice + 8KB B slice (4 x b128 per thread)
  auto stage = [&](int ks, int buf) {
    const int k0 = ks * 32;
#pragma unroll
    for (int i = 0; i < 2; ++i) {
      int c = tid + 256 * i;        // 0..511 -> 512 chunks of 16B
      int row = c >> 2;             // 0..127
      int seg = (c & 3) * 8;        // bf16 offset within 32-wide slice
      async_ld_b128(&lA[buf][row * 32 + seg], A + (size_t)(tileM + row) * K + k0 + seg);
      async_ld_b128(&lB[buf][row * 32 + seg], Bm + (size_t)(tileN + row) * K + k0 + seg);
    }
  };

  stage(0, 0);
  for (int ks = 0; ks < NK; ++ks) {
    const int cur = ks & 1;
    if (ks + 1 < NK) { stage(ks + 1, cur ^ 1); wait_async_le4(); }
    else             { wait_async_0(); }
    __syncthreads();   // all waves' async copies for `cur` are visible

    const bf16* Ab = &lA[cur][0];
    const bf16* Bb = &lB[cur][0];
    bf16x16 af[2], bf_[4];
#pragma unroll
    for (int mt = 0; mt < 2; ++mt) af[mt] = load_afrag(Ab + (size_t)(am + mt * 16) * 32, 32, lane);
#pragma unroll
    for (int nt = 0; nt < 4; ++nt) bf_[nt] = load_bfrag(Bb + (size_t)(bn + nt * 16) * 32, 32, lane);
#pragma unroll
    for (int mt = 0; mt < 2; ++mt)
#pragma unroll
      for (int nt = 0; nt < 4; ++nt)
        acc[mt][nt] = wmma_bf16(af[mt], bf_[nt], acc[mt][nt]);

    __syncthreads();   // done reading `cur` before it gets overwritten
  }

  const int half = lane >> 4;
  const int col  = lane & 15;
#pragma unroll
  for (int mt = 0; mt < 2; ++mt)
#pragma unroll
    for (int nt = 0; nt < 4; ++nt)
#pragma unroll
      for (int r = 0; r < 8; ++r) {
        int row = tileM + am + mt * 16 + r + 8 * half;  // C/D layout row
        int fcol = tileN + bn + nt * 16 + col;          // C/D layout col
        epi(row, fcol, acc[mt][nt][r]);
      }
}

// ---------- GEMM 1: qkv = x @ W_qkv^T, scatter into Q,K (B,H,T,Dh) and V^T (B,H,Dh,T) ----------

__global__ __launch_bounds__(256)
void k_gemm_qkv(const bf16* __restrict__ X, const bf16* __restrict__ W,
                bf16* __restrict__ Qo, bf16* __restrict__ Ko, bf16* __restrict__ VTo) {
  gemm_core(X, W, E_, [=](int token, int f, float v) {
    int which = f >> 10;        // 0=q 1=k 2=v
    int eidx  = f & 1023;
    int h = eidx >> 6, d = eidx & 63;
    int b = token >> 12, t = token & (T_ - 1);
    int bh = (b << 4) | h;
    bf16 val = (bf16)v;
    if (which == 0)      Qo[((size_t)bh * T_ + t) * DH_ + d] = val;
    else if (which == 1) Ko[((size_t)bh * T_ + t) * DH_ + d] = val;
    else                 VTo[((size_t)bh * DH_ + d) * T_ + t] = val;
  });
}

// ---------- GEMM 2: out = attn @ W_out^T (f32 output) ----------

__global__ __launch_bounds__(256)
void k_gemm_out(const bf16* __restrict__ A, const bf16* __restrict__ W,
                float* __restrict__ Out) {
  gemm_core(A, W, E_, [=](int token, int f, float v) {
    Out[(size_t)token * E_ + f] = v;
  });
}

// ---------- Attention: flash-style streaming softmax + FHN constant factor ----------
// grid = (4 qblocks, 8 chunks, 32 b*h); 8 waves/WG, each wave owns 16 query rows.

__global__ __launch_bounds__(256)
void k_attn(const bf16* __restrict__ Q, const bf16* __restrict__ Kb,
            const bf16* __restrict__ VT, bf16* __restrict__ Attn,
            const float* __restrict__ pa, const float* __restrict__ pb,
            const float* __restrict__ pdt) {
  __shared__ __align__(16) bf16 ldsP[8][16 * 32];  // per-wave P staging (C-layout -> A-layout)

  const int lane = threadIdx.x & 31;
  const int wave = threadIdx.x >> 5;
  const int half = lane >> 4;
  const int col  = lane & 15;
  const int ci = blockIdx.y;
  const int bh = blockIdx.z;
  const int b = bh >> 4;
  const int h = bh & 15;
  const int e = (ci + 1) * CHUNK_;                       // keys 0..e-1
  const int q0 = ci * CHUNK_ + blockIdx.x * 128 + wave * 16;

  const bf16* Qbase = Q + ((size_t)bh * T_ + q0) * DH_;
  const bf16* Kbase = Kb + (size_t)bh * T_ * DH_;
  const bf16* Vbase = VT + (size_t)bh * DH_ * T_;

  floatx8 o[4];
#pragma unroll
  for (int t = 0; t < 4; ++t)
#pragma unroll
    for (int r = 0; r < 8; ++r) o[t][r] = 0.0f;

  float m8[8], l8[8];
#pragma unroll
  for (int r = 0; r < 8; ++r) { m8[r] = -1e30f; l8[r] = 0.0f; }

  const float scale = 0.125f;  // 1/sqrt(64)
  bf16* pbuf = &ldsP[wave][0];

  for (int kb = 0; kb < e; kb += 32) {
    // S = (q_c @ K^T) * scale : two 16x16 tiles (32 keys)
    floatx8 s[2];
#pragma unroll
    for (int nt = 0; nt < 2; ++nt) {
      floatx8 sa;
#pragma unroll
      for (int r = 0; r < 8; ++r) sa[r] = 0.0f;
#pragma unroll
      for (int kd = 0; kd < DH_; kd += 32) {
        bf16x16 aq = load_afrag(Qbase + kd, DH_, lane);
        bf16x16 bk = load_bfrag(Kbase + (size_t)(kb + nt * 16) * DH_ + kd, DH_, lane);
        sa = wmma_bf16(aq, bk, sa);
      }
      s[nt] = sa * scale;
    }

    // online softmax stats
    float fac[8], psum[8];
#pragma unroll
    for (int r = 0; r < 8; ++r) {
      float mx = rmax16(fmaxf(s[0][r], s[1][r]));
      float mnew = fmaxf(m8[r], mx);
      fac[r] = __expf(m8[r] - mnew);
      m8[r] = mnew;
      psum[r] = 0.0f;
    }
    // P = exp(S - m), row sums, stage P into LDS in (row,key) order
#pragma unroll
    for (int nt = 0; nt < 2; ++nt)
#pragma unroll
      for (int r = 0; r < 8; ++r) {
        float pv = __expf(s[nt][r] - m8[r]);
        psum[r] += rsum16(pv);
        pbuf[(size_t)(r + 8 * half) * 32 + nt * 16 + col] = (bf16)pv;
      }
#pragma unroll
    for (int r = 0; r < 8; ++r) l8[r] = l8[r] * fac[r] + psum[r];
#pragma unroll
    for (int t = 0; t < 4; ++t)
#pragma unroll
      for (int r = 0; r < 8; ++r) o[t][r] *= fac[r];

    asm volatile("s_wait_dscnt 0" ::: "memory");  // LDS RAW: P store -> A-frag load

    bf16x16 pf = load_afrag(pbuf, 32, lane);
    // O += P @ V  (V^T layout: [dim][token], ld = T_)
#pragma unroll
    for (int t = 0; t < 4; ++t) {
      bf16x16 bv = load_bfrag(Vbase + (size_t)(t * 16) * T_ + kb, T_, lane);
      o[t] = wmma_bf16(pf, bv, o[t]);
    }
  }

  // FHN gating: attn_energy == 1 after softmax -> per-row constant factor
  float a_ = *pa, b_ = *pb, dt_ = *pdt;
  float gate = 1.0f / (1.0f + __expf(-5.0f));       // sigmoid((|1|-0.5)*10)
  float I = 0.1f + 0.9f * gate;
  float alpha = dt_ / 12.5f;
  float denom = 1.0f + alpha * b_;
  float vf = 0.0f, wf = 0.0f;
#pragma unroll
  for (int it = 0; it < 2; ++it) {
    float dv = vf - vf * vf * vf * (1.0f / 3.0f) - wf + I;
    float vn = vf + dt_ * dv;
    float wn = (wf + alpha * (vn + a_)) / denom;
    vf = fminf(fmaxf(vn, -3.0f), 3.0f);
    wf = fminf(fmaxf(wn, -3.0f), 3.0f);
  }
  float g2 = 1.0f / (1.0f + __expf(-vf));
  float c = 0.5f + 0.5f * g2;
  float factor = c / (c + 1e-8f);

  // write attn output (B,T,E) as bf16 for the final GEMM
#pragma unroll
  for (int t = 0; t < 4; ++t)
#pragma unroll
    for (int r = 0; r < 8; ++r) {
      int row = q0 + r + 8 * half;
      int d = t * 16 + col;
      float val = o[t][r] / l8[r] * factor;
      Attn[((size_t)b * T_ + row) * E_ + h * DH_ + d] = (bf16)val;
    }
}

// ---------- launch ----------

extern "C" void kernel_launch(void* const* d_in, const int* in_sizes, int n_in,
                              void* d_out, int out_size, void* d_ws, size_t ws_size,
                              hipStream_t stream) {
  (void)in_sizes; (void)n_in; (void)out_size; (void)ws_size;
  const float* x    = (const float*)d_in[0];
  // d_in[1] = spectral_basis (unused by reference)
  const float* Wqkv = (const float*)d_in[2];
  const float* Wout = (const float*)d_in[3];
  const float* pa   = (const float*)d_in[4];
  const float* pb   = (const float*)d_in[5];
  const float* pdt  = (const float*)d_in[6];
  float* out = (float*)d_out;

  char* ws = (char*)d_ws;
  size_t off = 0;
  const size_t NX  = (size_t)B_ * T_ * E_;      // 8,388,608
  const size_t NW1 = (size_t)3 * E_ * E_;       // 3,145,728
  const size_t NW2 = (size_t)E_ * E_;           // 1,048,576
  const size_t NQ  = (size_t)B_ * H_ * T_ * DH_;

  bf16* xb    = (bf16*)(ws + off); off += NX  * sizeof(bf16);
  bf16* wqkvb = (bf16*)(ws + off); off += NW1 * sizeof(bf16);
  bf16* woutb = (bf16*)(ws + off); off += NW2 * sizeof(bf16);
  bf16* qb    = (bf16*)(ws + off); off += NQ  * sizeof(bf16);
  bf16* kb    = (bf16*)(ws + off); off += NQ  * sizeof(bf16);
  bf16* vtb   = (bf16*)(ws + off); off += NQ  * sizeof(bf16);
  bf16* attnb = (bf16*)(ws + off); off += NX  * sizeof(bf16);

  k_cvt<<<dim3((unsigned)((NX  + 255) / 256)), 256, 0, stream>>>(x,    xb,    (int)NX);
  k_cvt<<<dim3((unsigned)((NW1 + 255) / 256)), 256, 0, stream>>>(Wqkv, wqkvb, (int)NW1);
  k_cvt<<<dim3((unsigned)((NW2 + 255) / 256)), 256, 0, stream>>>(Wout, woutb, (int)NW2);

  // M=8192, N=3072 -> 64 x 24 tiles of 128x128
  k_gemm_qkv<<<dim3(64, 24), 256, 0, stream>>>(xb, wqkvb, qb, kb, vtb);

  k_attn<<<dim3(4, 8, 32), 256, 0, stream>>>(qb, kb, vtb, attnb, pa, pb, pdt);

  // M=8192, N=1024 -> 64 x 8 tiles of 128x128
  k_gemm_out<<<dim3(64, 8), 256, 0, stream>>>(attnb, woutb, out);
}
